// MOE_ViTsmall_Top_1_20272245637148
// MI455X (gfx1250) — matrix operand order repaired and checked by reading, not compiled, not run on hardware
//
#include <hip/hip_runtime.h>

#define DEV __device__ __forceinline__

typedef __attribute__((ext_vector_type(16))) _Float16 v16h;
typedef __attribute__((ext_vector_type(8)))  float    v8f;

constexpr int E = 8, DL = 6, DIM = 512, H = 8, DH = 64, MLP = 512, NC = 10;
constexpr int S = 32, P = 4, PD = 48, NTOK = 65, NP = 80, BATCH = 128;

// ---------------- WMMA fragment helpers (gfx1250 wave32 layouts) ----------------
// A 16x32 f16: lane l holds row M = l&15; half j -> K = j + (l>>4)*16
DEV v16h ldA(const _Float16* a, int lda, int m0, int k0, int lane) {
  const _Float16* r0 = a + (m0 + (lane & 15)) * lda + k0 + ((lane >> 4) << 4);
  v16h r;
#pragma unroll
  for (int j = 0; j < 8; ++j) r[j] = r0[j];
#pragma unroll
  for (int j = 0; j < 8; ++j) r[j + 8] = r0[8 + j];
  return r;
}
// B 32x16 f16 from row-major [K][N] in LDS: lane l holds col N = l&15; half j -> K = j + (l>>4)*16
DEV v16h ldB_lds(const _Float16* b, int ldb, int k0, int n0, int lane) {
  const _Float16* c0 = b + (k0 + ((lane >> 4) << 4)) * ldb + n0 + (lane & 15);
  v16h r;
#pragma unroll
  for (int j = 0; j < 16; ++j) r[j] = c0[j * ldb];
  return r;
}
// B 32x16 from row-major fp32 [K][N] in global, cvt f32->f16 in registers
DEV v16h ldB_g32(const float* b, int ldb, int k0, int n0, int lane) {
  const float* c0 = b + (size_t)(k0 + ((lane >> 4) << 4)) * ldb + (n0 + (lane & 15));
  v16h r;
#pragma unroll
  for (int j = 0; j < 16; ++j) r[j] = (_Float16)c0[(size_t)j * ldb];
  return r;
}
// B = src^T where src is row-major [N][K] f16 in LDS (for q @ k^T)
DEV v16h ldBT_lds(const _Float16* b, int ldb, int k0, int n0, int lane) {
  const _Float16* r0 = b + (n0 + (lane & 15)) * ldb + k0 + ((lane >> 4) << 4);
  v16h r;
#pragma unroll
  for (int j = 0; j < 16; ++j) r[j] = r0[j];
  return r;
}
DEV v8f wmma(v16h a, v16h b, v8f c) {
  return __builtin_amdgcn_wmma_f32_16x16x32_f16(false, a, false, b, (short)0, c, false, false);
}
DEV float wave_sum(float v) {
#pragma unroll
  for (int m = 16; m >= 1; m >>= 1) v += __shfl_xor(v, m, 32);
  return v;
}
DEV float wave_max(float v) {
#pragma unroll
  for (int m = 16; m >= 1; m >>= 1) v = fmaxf(v, __shfl_xor(v, m, 32));
  return v;
}
// prefetch next weight K-panel into cache (global_prefetch_b8)
DEV void pf(const float* p) { __builtin_prefetch((const void*)p, 0, 1); }

// Async global->LDS copy of 16 bytes per lane (GLOBAL_LOAD_ASYNC_TO_LDS_B128, ASYNCcnt)
DEV void async_copy16(unsigned lds_off, const void* gptr) {
  unsigned long long ga = (unsigned long long)gptr;
  asm volatile("global_load_async_to_lds_b128 %0, %1, off"
               :: "v"(lds_off), "v"(ga) : "memory");
}
DEV void wait_async0() { asm volatile("s_wait_asynccnt 0" ::: "memory"); }

// LayerNorm(tres rows 0..64) * g + b -> Abuf (f16). Pad rows 65..79 stay zero.
DEV void ln_to_abuf(float (*tr)[DIM], _Float16 (*ab)[DIM],
                    const float* g, const float* bb, int wave, int lane) {
  for (int r = wave; r < NTOK; r += 8) {
    float vals[16]; float sum = 0.f;
#pragma unroll
    for (int i = 0; i < 16; ++i) { vals[i] = tr[r][lane + i * 32]; sum += vals[i]; }
    float mean = wave_sum(sum) * (1.f / DIM);
    float vs = 0.f;
#pragma unroll
    for (int i = 0; i < 16; ++i) { vals[i] -= mean; vs += vals[i] * vals[i]; }
    float rstd = rsqrtf(wave_sum(vs) * (1.f / DIM) + 1e-5f);
#pragma unroll
    for (int i = 0; i < 16; ++i) {
      int c = lane + i * 32;
      ab[r][c] = (_Float16)(vals[i] * rstd * g[c] + bb[c]);
    }
  }
}

// ---------------- Kernel 1: gate + top-1 routing ----------------
__global__ __launch_bounds__(256) void gate_kernel(const float* __restrict__ x,
                                                   const float* __restrict__ gw,
                                                   const float* __restrict__ gb,
                                                   int* __restrict__ expert) {
  const int b = blockIdx.x;
  const int wave = threadIdx.x >> 5, lane = threadIdx.x & 31;
  __shared__ float slog[E];
  const float* xb = x + (size_t)b * 3 * S * S;
  const float* we = gw + (size_t)wave * 3 * S * S;
  float s = 0.f;
  for (int i = lane; i < 3 * S * S; i += 32) s += xb[i] * we[i];
  s = wave_sum(s);
  if (lane == 0) slog[wave] = s + gb[wave];
  __syncthreads();
  if (threadIdx.x == 0) {
    int best = 0; float bv = slog[0];
    for (int k = 1; k < E; ++k) if (slog[k] > bv) { bv = slog[k]; best = k; }
    expert[b] = best;
  }
}

// ---------------- Kernel 2: full ViT forward for the routed expert ----------------
__global__ __launch_bounds__(256, 1)
void vit_kernel(const float* __restrict__ x,
                const float* __restrict__ pln1_g, const float* __restrict__ pln1_b,
                const float* __restrict__ patch_w, const float* __restrict__ patch_b,
                const float* __restrict__ pln2_g, const float* __restrict__ pln2_b,
                const float* __restrict__ cls_tok, const float* __restrict__ pos_emb,
                const float* __restrict__ aln_g, const float* __restrict__ aln_b,
                const float* __restrict__ qkv_w, const float* __restrict__ out_w,
                const float* __restrict__ out_b,
                const float* __restrict__ fln_g, const float* __restrict__ fln_b,
                const float* __restrict__ ff1_w, const float* __restrict__ ff1_b,
                const float* __restrict__ ff2_w, const float* __restrict__ ff2_b,
                const float* __restrict__ final_g, const float* __restrict__ final_b,
                const float* __restrict__ head_w, const float* __restrict__ head_b,
                const int* __restrict__ expert, float* __restrict__ out) {
  __shared__ __align__(16) float    tres[NTOK][DIM];   // fp32 residual, 130 KB (also async stage)
  __shared__ __align__(16) _Float16 Abuf[NP][DIM];     // f16 GEMM-A staging, 80 KB
  __shared__ __align__(16) _Float16 Braw[NP * DIM];    // f16 multi-purpose scratch, 80 KB
  __shared__ float fin[DIM];                            // final-LN row

  const int b = blockIdx.x;
  const int tid = threadIdx.x, wave = tid >> 5, lane = tid & 31;
  const int e = expert[b];

  // ---- phase 0a: async-DMA patch_w (fp32, 48x512 = 96 KB) into tres scratch region
  {
    const char* src = (const char*)(patch_w + (size_t)e * PD * DIM);
    unsigned ldsBase = (unsigned)(uintptr_t)&tres[0][0];   // low 32 bits = LDS offset
    for (int it = 0; it < (PD * DIM * 4) / (256 * 16); ++it) {
      unsigned boff = (unsigned)(it * 256 + tid) * 16u;
      async_copy16(ldsBase + boff, src + boff);
    }
  }
  // ---- phase 0b: zero A staging (pad rows must be zero)
  for (int i = tid; i < NP * DIM; i += 256) (&Abuf[0][0])[i] = (_Float16)0.f;
  wait_async0();
  __syncthreads();

  // ---- phase 1a: convert staged fp32 patch_w -> f16 Braw [64][512] (rows 48..63 zero)
  {
    const float* stage = &tres[0][0];
    for (int i = tid; i < 64 * DIM; i += 256) {
      int r = i >> 9, c = i & (DIM - 1);
      Braw[i] = (r < PD) ? (_Float16)stage[r * DIM + c] : (_Float16)0.f;
    }
  }
  // ---- phase 1b: patch extraction + LN1 -> Abuf rows 1..64, cols 0..47 (48..63 zero)
  const float* g1 = pln1_g + e * PD;
  const float* b1 = pln1_b + e * PD;
  for (int p = wave; p < 64; p += 8) {
    int ph = p >> 3, pwid = p & 7;
    int j0 = lane, j1 = lane + 32;
    float xv0, xv1 = 0.f;
    { int c = j0 % 3, pp = j0 / 3, p1 = pp >> 2, p2 = pp & 3;
      xv0 = x[(((size_t)b * 3 + c) * S + (ph * P + p1)) * S + (pwid * P + p2)]; }
    if (j1 < PD) {
      int c = j1 % 3, pp = j1 / 3, p1 = pp >> 2, p2 = pp & 3;
      xv1 = x[(((size_t)b * 3 + c) * S + (ph * P + p1)) * S + (pwid * P + p2)];
    }
    float mean = wave_sum(xv0 + xv1) * (1.f / PD);
    float d0 = xv0 - mean, d1 = (j1 < PD) ? (xv1 - mean) : 0.f;
    float rstd = rsqrtf(wave_sum(d0 * d0 + d1 * d1) * (1.f / PD) + 1e-5f);
    Abuf[p + 1][j0] = (_Float16)(d0 * rstd * g1[j0] + b1[j0]);
    if (j1 < PD) Abuf[p + 1][j1] = (_Float16)(d1 * rstd * g1[j1] + b1[j1]);
  }
  __syncthreads();

  // ---- patch-embed GEMM [80][512] (K=64, zero-padded) + patch_b -> tres rows 1..64
  const float* pb = patch_b + e * DIM;
  for (int t = wave; t < 5 * 32; t += 8) {
    int mt = t % 5, nt = t / 5, m0 = mt * 16, n0 = nt * 16;
    v8f acc = {};
    for (int k0 = 0; k0 < 64; k0 += 32)
      acc = wmma(ldA(&Abuf[0][0], DIM, m0, k0, lane), ldB_lds(Braw, DIM, k0, n0, lane), acc);
    int col = n0 + (lane & 15), mrow = m0 + ((lane >> 4) << 3);
    if (mt < 4) {                       // rows fully < 64 except row 0 of tile 0
#pragma unroll
      for (int i = 0; i < 8; ++i) {
        if (mt == 0 && i == 0 && lane < 16) continue;   // token 0 = cls, set later
        tres[mrow + i][col] = acc[i] + pb[col];
      }
    } else if (lane < 16) {             // tile 4: only row 64 valid (element 0, lanes 0-15)
      tres[64][col] = acc[0] + pb[col];
    }
  }
  __syncthreads();

  // ---- LN2 on patch tokens, add pos_emb; token 0 = cls + pos
  {
    const float* g2 = pln2_g + e * DIM; const float* b2 = pln2_b + e * DIM;
    const float* pe = pos_emb + (size_t)e * NTOK * DIM;
    const float* ct = cls_tok + e * DIM;
    for (int r = wave; r < NTOK; r += 8) {
      if (r == 0) {
        for (int c = lane; c < DIM; c += 32) tres[0][c] = ct[c] + pe[c];
      } else {
        float vals[16]; float sum = 0.f;
#pragma unroll
        for (int i = 0; i < 16; ++i) { vals[i] = tres[r][lane + i * 32]; sum += vals[i]; }
        float mean = wave_sum(sum) * (1.f / DIM);
        float vs = 0.f;
#pragma unroll
        for (int i = 0; i < 16; ++i) { vals[i] -= mean; vs += vals[i] * vals[i]; }
        float rstd = rsqrtf(wave_sum(vs) * (1.f / DIM) + 1e-5f);
#pragma unroll
        for (int i = 0; i < 16; ++i) {
          int c = lane + i * 32;
          tres[r][c] = vals[i] * rstd * g2[c] + b2[c] + pe[r * DIM + c];
        }
      }
    }
  }
  __syncthreads();

  // ---- transformer layers
  _Float16* qbuf = Braw;                          // [80][64]
  _Float16* kbuf = Braw + 80 * 64;                // [80][64]
  _Float16* vbuf = Braw + 2 * 80 * 64;            // [96][64] (rows 80..95 zero)
  _Float16* sbuf = Braw + 2 * 80 * 64 + 96 * 64;  // [80][96]
  _Float16* obuf = sbuf + 80 * 96;                // [80][64]

  for (int l = 0; l < DL; ++l) {
    const size_t li = (size_t)e * DL + l;
    ln_to_abuf(tres, Abuf, aln_g + li * DIM, aln_b + li * DIM, wave, lane);
    __syncthreads();

    const float* Wqkv = qkv_w + li * DIM * (3 * H * DH);
    const float* Wout = out_w + li * (H * DH) * DIM;

    for (int h = 0; h < H; ++h) {
      for (int i = tid; i < 16 * 64; i += 256) vbuf[80 * 64 + i] = (_Float16)0.f;
      // q,k,v projections: 60 tiles of [80][64], K=512
      for (int t = wave; t < 60; t += 8) {
        int which = t / 20, sub = t % 20;
        int mt = sub % 5, nt = sub / 5, m0 = mt * 16, n0 = nt * 16;
        int gcol = which * DIM + h * DH + n0;
        v8f acc = {};
        for (int k0 = 0; k0 < DIM; k0 += 32) {
          if (k0 + 32 < DIM)
            pf(Wqkv + (size_t)(k0 + 32 + ((lane >> 4) << 4)) * (3 * H * DH) + gcol + (lane & 15));
          acc = wmma(ldA(&Abuf[0][0], DIM, m0, k0, lane),
                     ldB_g32(Wqkv, 3 * H * DH, k0, gcol, lane), acc);
        }
        _Float16* dst = (which == 0) ? qbuf : (which == 1) ? kbuf : vbuf;
        int col = n0 + (lane & 15), mrow = m0 + ((lane >> 4) << 3);
#pragma unroll
        for (int i = 0; i < 8; ++i) dst[(mrow + i) * DH + col] = (_Float16)acc[i];
      }
      __syncthreads();
      // scores = (q @ k^T) / 8 : [80][80]
      for (int t = wave; t < 25; t += 8) {
        int mt = t % 5, nt = t / 5, m0 = mt * 16, n0 = nt * 16;
        v8f acc = {};
        for (int k0 = 0; k0 < DH; k0 += 32)
          acc = wmma(ldA(qbuf, DH, m0, k0, lane), ldBT_lds(kbuf, DH, k0, n0, lane), acc);
        int col = n0 + (lane & 15), mrow = m0 + ((lane >> 4) << 3);
#pragma unroll
        for (int i = 0; i < 8; ++i) sbuf[(mrow + i) * 96 + col] = (_Float16)(acc[i] * 0.125f);
      }
      __syncthreads();
      // masked softmax over valid 65 columns; pad cols written 0
      for (int r = wave; r < NP; r += 8) {
        float c0 = (float)sbuf[r * 96 + lane];
        float c1 = (lane + 32 < NTOK) ? (float)sbuf[r * 96 + lane + 32] : -1e30f;
        float c2 = (lane + 64 < NTOK) ? (float)sbuf[r * 96 + lane + 64] : -1e30f;
        float mx = wave_max(fmaxf(c0, fmaxf(c1, c2)));
        float e0 = expf(c0 - mx);
        float e1 = (lane + 32 < NTOK) ? expf(c1 - mx) : 0.f;
        float e2 = (lane + 64 < NTOK) ? expf(c2 - mx) : 0.f;
        float inv = 1.f / wave_sum(e0 + e1 + e2);
        sbuf[r * 96 + lane]      = (_Float16)(e0 * inv);
        sbuf[r * 96 + lane + 32] = (_Float16)(e1 * inv);
        sbuf[r * 96 + lane + 64] = (_Float16)(e2 * inv);
      }
      __syncthreads();
      // o = softmax(s) @ v : [80][64], K=96 (zero-padded)
      for (int t = wave; t < 20; t += 8) {
        int mt = t % 5, nt = t / 5, m0 = mt * 16, n0 = nt * 16;
        v8f acc = {};
        for (int k0 = 0; k0 < 96; k0 += 32)
          acc = wmma(ldA(sbuf, 96, m0, k0, lane), ldB_lds(vbuf, DH, k0, n0, lane), acc);
        int col = n0 + (lane & 15), mrow = m0 + ((lane >> 4) << 3);
#pragma unroll
        for (int i = 0; i < 8; ++i) obuf[(mrow + i) * DH + col] = (_Float16)acc[i];
      }
      __syncthreads();
      // residual accumulate: tres += o_h @ Wout[h] (fp32 residual rides the WMMA C fragment)
      const float* Wh = Wout + (size_t)(h * DH) * DIM;
      for (int t = wave; t < 5 * 32; t += 8) {
        int mt = t % 5, nt = t / 5, m0 = mt * 16, n0 = nt * 16;
        int col = n0 + (lane & 15), mrow = m0 + ((lane >> 4) << 3);
        v8f acc;
        if (mt < 4) {
#pragma unroll
          for (int i = 0; i < 8; ++i) acc[i] = tres[mrow + i][col];
        } else {
#pragma unroll
          for (int i = 0; i < 8; ++i) acc[i] = 0.f;
          if (lane < 16) acc[0] = tres[64][col];
        }
        for (int k0 = 0; k0 < DH; k0 += 32) {
          if (k0 + 32 < DH)
            pf(Wh + (size_t)(k0 + 32 + ((lane >> 4) << 4)) * DIM + n0 + (lane & 15));
          acc = wmma(ldA(obuf, DH, m0, k0, lane), ldB_g32(Wh, DIM, k0, n0, lane), acc);
        }
        if (mt < 4) {
#pragma unroll
          for (int i = 0; i < 8; ++i) tres[mrow + i][col] = acc[i];
        } else if (lane < 16) {
          tres[64][col] = acc[0];
        }
      }
      __syncthreads();
    }
    // + out_b
    const float* ob = out_b + li * DIM;
    for (int i = tid; i < NTOK * DIM; i += 256) (&tres[0][0])[i] += ob[i & (DIM - 1)];
    __syncthreads();

    // ---- MLP
    ln_to_abuf(tres, Abuf, fln_g + li * DIM, fln_b + li * DIM, wave, lane);
    __syncthreads();
    const float* W1 = ff1_w + li * DIM * MLP;
    const float* bb1 = ff1_b + li * MLP;
    for (int t = wave; t < 5 * (MLP / 16); t += 8) {
      int mt = t % 5, nt = t / 5, m0 = mt * 16, n0 = nt * 16;
      v8f acc = {};
      for (int k0 = 0; k0 < DIM; k0 += 32) {
        if (k0 + 32 < DIM)
          pf(W1 + (size_t)(k0 + 32 + ((lane >> 4) << 4)) * MLP + n0 + (lane & 15));
        acc = wmma(ldA(&Abuf[0][0], DIM, m0, k0, lane), ldB_g32(W1, MLP, k0, n0, lane), acc);
      }
      int col = n0 + (lane & 15), mrow = m0 + ((lane >> 4) << 3);
#pragma unroll
      for (int i = 0; i < 8; ++i) {
        float v = acc[i] + bb1[col];
        v = 0.5f * v * (1.f + erff(v * 0.70710678118f));   // exact GELU
        Braw[(mrow + i) * MLP + col] = (_Float16)v;
      }
    }
    __syncthreads();
    const float* W2 = ff2_w + li * MLP * DIM;
    for (int t = wave; t < 5 * 32; t += 8) {
      int mt = t % 5, nt = t / 5, m0 = mt * 16, n0 = nt * 16;
      int col = n0 + (lane & 15), mrow = m0 + ((lane >> 4) << 3);
      v8f acc;
      if (mt < 4) {
#pragma unroll
        for (int i = 0; i < 8; ++i) acc[i] = tres[mrow + i][col];
      } else {
#pragma unroll
        for (int i = 0; i < 8; ++i) acc[i] = 0.f;
        if (lane < 16) acc[0] = tres[64][col];
      }
      for (int k0 = 0; k0 < MLP; k0 += 32) {
        if (k0 + 32 < MLP)
          pf(W2 + (size_t)(k0 + 32 + ((lane >> 4) << 4)) * DIM + n0 + (lane & 15));
        acc = wmma(ldA(Braw, MLP, m0, k0, lane), ldB_g32(W2, DIM, k0, n0, lane), acc);
      }
      if (mt < 4) {
#pragma unroll
        for (int i = 0; i < 8; ++i) tres[mrow + i][col] = acc[i];
      } else if (lane < 16) {
        tres[64][col] = acc[0];
      }
    }
    __syncthreads();
    const float* bb2 = ff2_b + li * DIM;
    for (int i = tid; i < NTOK * DIM; i += 256) (&tres[0][0])[i] += bb2[i & (DIM - 1)];
    __syncthreads();
  }

  // ---- final LN on cls token + head
  const float* fg = final_g + e * DIM; const float* fb = final_b + e * DIM;
  if (wave == 0) {
    float vals[16]; float sum = 0.f;
#pragma unroll
    for (int i = 0; i < 16; ++i) { vals[i] = tres[0][lane + i * 32]; sum += vals[i]; }
    float mean = wave_sum(sum) * (1.f / DIM);
    float vs = 0.f;
#pragma unroll
    for (int i = 0; i < 16; ++i) { vals[i] -= mean; vs += vals[i] * vals[i]; }
    float rstd = rsqrtf(wave_sum(vs) * (1.f / DIM) + 1e-5f);
#pragma unroll
    for (int i = 0; i < 16; ++i) {
      int c = lane + i * 32;
      fin[c] = vals[i] * rstd * fg[c] + fb[c];
    }
  }
  __syncthreads();
  if (tid < NC) {
    const float* hw = head_w + (size_t)e * DIM * NC;
    float acc = head_b[e * NC + tid];
    for (int k = 0; k < DIM; ++k) acc += fin[k] * hw[k * NC + tid];
    out[(size_t)b * NC + tid] = acc;
  }
}

extern "C" void kernel_launch(void* const* d_in, const int* in_sizes, int n_in,
                              void* d_out, int out_size, void* d_ws, size_t ws_size,
                              hipStream_t stream) {
  (void)in_sizes; (void)n_in; (void)out_size; (void)ws_size;
  const float* x       = (const float*)d_in[0];
  const float* gate_w  = (const float*)d_in[1];
  const float* gate_b  = (const float*)d_in[2];
  const float* pln1_g  = (const float*)d_in[3];
  const float* pln1_b  = (const float*)d_in[4];
  const float* patch_w = (const float*)d_in[5];
  const float* patch_b = (const float*)d_in[6];
  const float* pln2_g  = (const float*)d_in[7];
  const float* pln2_b  = (const float*)d_in[8];
  const float* cls_tok = (const float*)d_in[9];
  const float* pos_emb = (const float*)d_in[10];
  const float* aln_g   = (const float*)d_in[11];
  const float* aln_b   = (const float*)d_in[12];
  const float* qkv_w   = (const float*)d_in[13];
  const float* out_w   = (const float*)d_in[14];
  const float* out_b   = (const float*)d_in[15];
  const float* fln_g   = (const float*)d_in[16];
  const float* fln_b   = (const float*)d_in[17];
  const float* ff1_w   = (const float*)d_in[18];
  const float* ff1_b   = (const float*)d_in[19];
  const float* ff2_w   = (const float*)d_in[20];
  const float* ff2_b   = (const float*)d_in[21];
  const float* final_g = (const float*)d_in[22];
  const float* final_b = (const float*)d_in[23];
  const float* head_w  = (const float*)d_in[24];
  const float* head_b  = (const float*)d_in[25];

  int* expert = (int*)d_ws;
  gate_kernel<<<dim3(BATCH), dim3(256), 0, stream>>>(x, gate_w, gate_b, expert);
  vit_kernel<<<dim3(BATCH), dim3(256), 0, stream>>>(
      x, pln1_g, pln1_b, patch_w, patch_b, pln2_g, pln2_b, cls_tok, pos_emb,
      aln_g, aln_b, qkv_w, out_w, out_b, fln_g, fln_b,
      ff1_w, ff1_b, ff2_w, ff2_b, final_g, final_b, head_w, head_b,
      expert, (float*)d_out);
}
